// SkExpKernelAttentionResid_39393440039235
// MI455X (gfx1250) — matrix operand order repaired and checked by reading, not compiled, not run on hardware
//
#include <hip/hip_runtime.h>

// ---------------------------------------------------------------------------
// SkExpKernelAttentionResid for MI455X (gfx1250, wave32, WMMA)
//   Shapes (fixed by reference): B*H = 32, S = 4096, D = 64, NSQ = NSK = 2048
//   out[b,h,idxq[s],:] += 4 * (Qsm @ (Ksm^T Vs)) + 2 * Vs     (scatter-add)
//   Ksm[s,d] = exp(Qsm[s,d]) / colsum[d]   (col-softmax, max-shift removable)
// ---------------------------------------------------------------------------

typedef __attribute__((ext_vector_type(16))) _Float16 v16h;
typedef __attribute__((ext_vector_type(8)))  float    v8f;

#define BH        32
#define S_TOT     4096
#define DD        64
#define NSQ       2048
#define SPLITS    8
#define ROWS_WG   (NSQ / SPLITS)      // 256 sampled rows per workgroup
#define BLK       32                  // rows per K-block (WMMA K for f16)
#define NBLK      (ROWS_WG / BLK)     // 8
#define LDH       72                  // padded f16 LDS row stride
#define STATS_PER_BH (DD * DD + DD)   // N (64x64) + colsum (64)

union Frag16 { v16h v; _Float16 h[16]; };

__device__ __forceinline__ float redMax8(float x) {
  x = fmaxf(x, __shfl_xor(x, 1, 8));
  x = fmaxf(x, __shfl_xor(x, 2, 8));
  x = fmaxf(x, __shfl_xor(x, 4, 8));
  return x;
}
__device__ __forceinline__ float redSum8(float x) {
  x += __shfl_xor(x, 1, 8);
  x += __shfl_xor(x, 2, 8);
  x += __shfl_xor(x, 4, 8);
  return x;
}

// Gather one Q row (8 threads/row, 8 floats/thread), fused row-softmax.
// Returns the 8 softmax values for this thread's column slice.
__device__ __forceinline__ void rowSoftmax8(const float* __restrict__ qrow,
                                            float o[8]) {
  const float4* qp = (const float4*)qrow;
  float4 q0 = qp[0], q1 = qp[1];
  float m = fmaxf(fmaxf(fmaxf(q0.x, q0.y), fmaxf(q0.z, q0.w)),
                  fmaxf(fmaxf(q1.x, q1.y), fmaxf(q1.z, q1.w)));
  m = redMax8(m);
  o[0] = __expf(q0.x - m); o[1] = __expf(q0.y - m);
  o[2] = __expf(q0.z - m); o[3] = __expf(q0.w - m);
  o[4] = __expf(q1.x - m); o[5] = __expf(q1.y - m);
  o[6] = __expf(q1.z - m); o[7] = __expf(q1.w - m);
  float s = o[0] + o[1] + o[2] + o[3] + o[4] + o[5] + o[6] + o[7];
  s = redSum8(s);
  float inv = 1.0f / s;
#pragma unroll
  for (int i = 0; i < 8; ++i) o[i] *= inv;
}

__global__ void zeroK(float* __restrict__ p, size_t n) {
  size_t i = (size_t)blockIdx.x * blockDim.x + threadIdx.x;
  size_t st = (size_t)gridDim.x * blockDim.x;
  for (; i < n; i += st) p[i] = 0.0f;
}

// ---------------------------------------------------------------------------
// Stage A: per (bh, split) accumulate N[d,e] = sum_s exp(Qsm[s,d]) * Vs[s,e]
//          and colsum[d] = sum_s exp(Qsm[s,d]) into ws (f32 atomics).
//          GEMM done with v_wmma_f32_16x16x32_f16 (A = E^T tile, B = Vs tile).
// ---------------------------------------------------------------------------
__global__ __launch_bounds__(256)
void stageA(const float* __restrict__ Q, const float* __restrict__ V,
            const int* __restrict__ idxq, const int* __restrict__ idxk,
            float* __restrict__ wsf) {
  __shared__ _Float16 Ebuf[BLK][LDH];   // exp(row-softmax(Q)) block, f16
  __shared__ _Float16 Vbuf[BLK][LDH];   // gathered V block, f16
  __shared__ float    csum[DD];

  const int bh    = blockIdx.x;
  const int split = blockIdx.y;
  const int tid   = threadIdx.x;
  const int lane  = tid & 31;
  const int wave  = tid >> 5;
  const int row   = tid >> 3;           // 0..31 (8 threads per row)
  const int cpart = tid & 7;            // 8-col slice id
  const size_t bhBase = (size_t)bh * S_TOT * DD;

  if (tid < DD) csum[tid] = 0.0f;

  // wave handles tiles t = 2w, 2w+1 of the 4x4 tile grid; shared d-tile.
  const int d0  = ((2 * wave) >> 2) * 16;
  const int ea0 = ((2 * wave) & 3) * 16;
  const int eb0 = ((2 * wave + 1) & 3) * 16;
  v8f acc0 = {}; v8f acc1 = {};

  __syncthreads();

  for (int blk = 0; blk < NBLK; ++blk) {
    const int sbase = split * ROWS_WG + blk * BLK;
    {
      const int sq = idxq[sbase + row];
      float sm[8];
      rowSoftmax8(Q + bhBase + (size_t)sq * DD + cpart * 8, sm);
      _Float16* ep = &Ebuf[row][cpart * 8];
#pragma unroll
      for (int i = 0; i < 8; ++i) ep[i] = (_Float16)__expf(sm[i]);

      const int sk = idxk[sbase + row];
      const float4* vp = (const float4*)(V + bhBase + (size_t)sk * DD + cpart * 8);
      float4 v0 = vp[0], v1 = vp[1];
      _Float16* vep = &Vbuf[row][cpart * 8];
      vep[0] = (_Float16)v0.x; vep[1] = (_Float16)v0.y;
      vep[2] = (_Float16)v0.z; vep[3] = (_Float16)v0.w;
      vep[4] = (_Float16)v1.x; vep[5] = (_Float16)v1.y;
      vep[6] = (_Float16)v1.z; vep[7] = (_Float16)v1.w;
    }
    __syncthreads();

    if (tid < DD) {                      // column-sum partials of E
      float cs = 0.0f;
#pragma unroll
      for (int r = 0; r < BLK; ++r) cs += (float)Ebuf[r][tid];
      csum[tid] += cs;
    }

    // Build WMMA fragments per ISA wave32 layouts (CDNA5 ISA §7.12.2).
    Frag16 a, b0, b1;
    const int mn  = lane & 15;
    const int kbA = (lane >> 4) * 8;
    const int kbB = (lane >> 4) * 16;
#pragma unroll
    for (int i = 0; i < 16; ++i) {
      const int kA = kbA + (i < 8 ? i : i + 8);
      a.h[i]  = Ebuf[kA][d0 + mn];       // A[m=d][k=s] = E[s][d]  (E^T)
      const int kB = kbA + (i < 8 ? i : i + 8);
      b0.h[i] = Vbuf[kB][ea0 + mn];      // B[k=s][n=e]
      b1.h[i] = Vbuf[kB][eb0 + mn];
    }
    acc0 = __builtin_amdgcn_wmma_f32_16x16x32_f16(false, a.v, false, b0.v,
                                                  (short)0, acc0, false, false);
    acc1 = __builtin_amdgcn_wmma_f32_16x16x32_f16(false, a.v, false, b1.v,
                                                  (short)0, acc1, false, false);
    __syncthreads();
  }

  // Reduce partials into ws: N (64x64) then colsum (64), per bh.
  float* Nmat = wsf + (size_t)bh * STATS_PER_BH;
  float* cs   = Nmat + DD * DD;
  const int n  = lane & 15;
  const int mo = (lane >> 4) * 8;
#pragma unroll
  for (int r = 0; r < 8; ++r) {
    const int m = r + mo;
    atomicAdd(&Nmat[(d0 + m) * DD + (ea0 + n)], acc0[r]);
    atomicAdd(&Nmat[(d0 + m) * DD + (eb0 + n)], acc1[r]);
  }
  if (tid < DD) atomicAdd(&cs[tid], csum[tid]);
}

// ---------------------------------------------------------------------------
// Stage B: dot = N / colsum (f16 in LDS), recompute Qsm per 32-row block,
//          ac = Qsm @ dot via WMMA (K=64 -> 2 steps), then
//          out[idxq[s]] += 4*ac + 2*V[idxk[s]]  (f32 global atomics).
// ---------------------------------------------------------------------------
__global__ __launch_bounds__(256)
void stageB(const float* __restrict__ Q, const float* __restrict__ V,
            const int* __restrict__ idxq, const int* __restrict__ idxk,
            const float* __restrict__ wsf, float* __restrict__ out) {
  __shared__ _Float16 Dbuf[DD][LDH];    // dot matrix, f16
  __shared__ _Float16 Abuf[BLK][LDH];   // Qsm block, f16

  const int bh    = blockIdx.x;
  const int split = blockIdx.y;
  const int tid   = threadIdx.x;
  const int lane  = tid & 31;
  const int wave  = tid >> 5;
  const size_t bhBase = (size_t)bh * S_TOT * DD;

  const float* Nmat = wsf + (size_t)bh * STATS_PER_BH;
  const float* cs   = Nmat + DD * DD;

  for (int i = tid; i < DD * DD; i += 256) {
    const int d = i >> 6, e = i & 63;
    Dbuf[d][e] = (_Float16)(Nmat[i] / cs[d]);
  }
  __syncthreads();

  const int row   = tid >> 3;
  const int cpart = tid & 7;
  const int m0 = (wave >> 2) * 16;      // 2 m-tiles x 4 e-tiles = 8 waves
  const int e0 = (wave & 3) * 16;

  for (int blk = 0; blk < NBLK; ++blk) {
    const int sbase = split * ROWS_WG + blk * BLK;
    {
      const int sq = idxq[sbase + row];
      float sm[8];
      rowSoftmax8(Q + bhBase + (size_t)sq * DD + cpart * 8, sm);
      _Float16* ap = &Abuf[row][cpart * 8];
#pragma unroll
      for (int i = 0; i < 8; ++i) ap[i] = (_Float16)sm[i];
    }
    __syncthreads();

    v8f acc = {};
    const int mn  = lane & 15;
    const int kbA = (lane >> 4) * 8;
    const int kbB = (lane >> 4) * 16;
#pragma unroll
    for (int kk = 0; kk < 2; ++kk) {    // K = 64 = 2 x 32
      Frag16 a, b;
#pragma unroll
      for (int i = 0; i < 16; ++i) {
        const int kA = kk * 32 + kbA + (i < 8 ? i : i + 8);
        a.h[i] = Abuf[m0 + mn][kA];     // A[m=s][k=d] = Qsm[s][d]
        const int kB = kk * 32 + kbA + (i < 8 ? i : i + 8);
        b.h[i] = Dbuf[kB][e0 + mn];     // B[k=d][n=e] = dot[d][e]
      }
      acc = __builtin_amdgcn_wmma_f32_16x16x32_f16(false, a.v, false, b.v,
                                                   (short)0, acc, false, false);
    }

    const int n  = lane & 15;
    const int mo = (lane >> 4) * 8;
#pragma unroll
    for (int r = 0; r < 8; ++r) {
      const int sl = m0 + r + mo;       // local sampled row 0..31
      const int sg = sbase + sl;
      const int orow = idxq[sg];
      const int vrow = idxk[sg];
      const float val = 4.0f * acc[r]
                      + 2.0f * V[bhBase + (size_t)vrow * DD + e0 + n];
      atomicAdd(&out[bhBase + (size_t)orow * DD + e0 + n], val);
    }
    __syncthreads();
  }
}

// ---------------------------------------------------------------------------
extern "C" void kernel_launch(void* const* d_in, const int* in_sizes, int n_in,
                              void* d_out, int out_size, void* d_ws, size_t ws_size,
                              hipStream_t stream) {
  (void)in_sizes; (void)n_in; (void)out_size; (void)ws_size;
  const float* Q    = (const float*)d_in[0];
  // d_in[1] (K tensor) and d_in[3] (mask) are unused by the reference.
  const float* V    = (const float*)d_in[2];
  const int*   idxq = (const int*)d_in[4];
  const int*   idxk = (const int*)d_in[5];
  float* out = (float*)d_out;
  float* wsf = (float*)d_ws;

  zeroK<<<2048, 256, 0, stream>>>(out, (size_t)BH * S_TOT * DD);
  zeroK<<<64,   256, 0, stream>>>(wsf, (size_t)BH * STATS_PER_BH);

  dim3 grid(BH, SPLITS);
  stageA<<<grid, 256, 0, stream>>>(Q, V, idxq, idxk, wsf);
  stageB<<<grid, 256, 0, stream>>>(Q, V, idxq, idxk, wsf, out);
}
